// PFGCNLayer_1864015806534
// MI455X (gfx1250) — matrix operand order repaired and checked by reading, not compiled, not run on hardware
//
#include <hip/hip_runtime.h>
#include <hip/hip_bf16.h>
#include <math.h>

typedef __attribute__((ext_vector_type(16))) _Float16 v16h;
typedef __attribute__((ext_vector_type(8)))  _Float16 v8h;
typedef __attribute__((ext_vector_type(4)))  _Float16 v4h;
typedef __attribute__((ext_vector_type(8)))  float    v8f;

#define TILE_M 64
#define TILE_N 64
#define TILE_K 32

union AFrag { v16h v; v8h h[2]; };

// ---- staging helpers (clamped, unconditional loads; garbage rows/cols are
//      masked at the final predicated store) ----
__device__ __forceinline__ void load_a_tile(const float* __restrict__ A,
                                            int rowBase, int M, int K, int k0,
                                            int tid, float4 out[4]) {
    #pragma unroll
    for (int t = 0; t < 4; ++t) {
        int idx = tid + t * 128;
        int r  = idx >> 3;               // 8 float4 per row
        int c0 = (idx & 7) << 2;
        int gr = rowBase + r;
        if (gr >= M) gr = M - 1;
        out[t] = *(const float4*)&A[(long)gr * K + (k0 + c0)];
    }
}

__device__ __forceinline__ void load_b_tile(const float* __restrict__ W,
                                            int colBase, int N, int k0,
                                            int tid, float4 out[4]) {
    #pragma unroll
    for (int t = 0; t < 4; ++t) {
        int idx = tid + t * 128;
        int r  = idx >> 4;               // 16 float4 per row
        int c0 = (idx & 15) << 2;
        int gc = colBase + c0;
        if (gc + 3 >= N) gc = 0;
        out[t] = *(const float4*)&W[(long)(k0 + r) * N + gc];
    }
}

// C[M,N] = A[M,K] @ W[K,N] + bias[N]; fp32 in/out, f16 WMMA with f32 accumulate.
// Requires K % 32 == 0 and N % 4 == 0 (holds: K=256, N in {256,40}).
__global__ __launch_bounds__(128) void gemm_bias_wmma(
    const float* __restrict__ A, const float* __restrict__ W,
    const float* __restrict__ bias, float* __restrict__ C,
    int M, int K, int N)
{
    // A tile, row stride 40 halves (80B): rows 16B-aligned, conflict-free b128 reads
    __shared__ __align__(16) _Float16 As[TILE_M][TILE_K + 8];
    // B tile pre-swizzled into WMMA fragment layout:
    //   Bsw[j][lane][k]  (lane = (kGroup<<4)|n15, k = K%16), lane stride 24 halves (48B)
    __shared__ __align__(16) _Float16 Bsw[4][32][24];

    const int tid     = threadIdx.x;
    const int lane    = tid & 31;
    const int wave    = tid >> 5;          // 0..3
    const int rowBase = blockIdx.x * TILE_M;
    const int colBase = blockIdx.y * TILE_N;

    v8f acc[4];
    #pragma unroll
    for (int j = 0; j < 4; ++j)
        #pragma unroll
        for (int v = 0; v < 8; ++v) acc[j][v] = 0.0f;

    // Register-pipelined staging: tile k0 already in regs when the loop body runs.
    float4 aCur[4], bCur[4];
    load_a_tile(A, rowBase, M, K, 0, tid, aCur);
    load_b_tile(W, colBase, N, 0, tid, bCur);

    for (int k0 = 0; k0 < K; k0 += TILE_K) {
        // ---- convert + store current tile regs to LDS ----
        #pragma unroll
        for (int t = 0; t < 4; ++t) {
            int idx = tid + t * 128;
            int r  = idx >> 3;
            int c0 = (idx & 7) << 2;
            v4h hv;
            hv[0] = (_Float16)aCur[t].x; hv[1] = (_Float16)aCur[t].y;
            hv[2] = (_Float16)aCur[t].z; hv[3] = (_Float16)aCur[t].w;
            *(v4h*)&As[r][c0] = hv;
        }
        #pragma unroll
        for (int t = 0; t < 4; ++t) {
            int idx = tid + t * 128;
            int r  = idx >> 4;
            int c0 = (idx & 15) << 2;
            int j  = c0 >> 4;
            int lb = ((r >> 4) << 4) | (c0 & 15);
            int rr = r & 15;
            Bsw[j][lb + 0][rr] = (_Float16)bCur[t].x;
            Bsw[j][lb + 1][rr] = (_Float16)bCur[t].y;
            Bsw[j][lb + 2][rr] = (_Float16)bCur[t].z;
            Bsw[j][lb + 3][rr] = (_Float16)bCur[t].w;
        }

        // ---- issue next tile's loads; they stay outstanding across the WMMAs ----
        const bool more = (k0 + TILE_K) < K;
        float4 aNxt[4], bNxt[4];
        if (more) {
            load_a_tile(A, rowBase, M, K, k0 + TILE_K, tid, aNxt);
            load_b_tile(W, colBase, N, k0 + TILE_K, tid, bNxt);
        }
        // Warm L2 two tiles ahead (global_prefetch_b8)
        if (k0 + 2 * TILE_K < K) {
            int pr = rowBase + (tid >> 1);
            if (pr >= M) pr = M - 1;
            __builtin_prefetch(&A[(long)pr * K + (k0 + 2 * TILE_K)], 0, 3);
            __builtin_prefetch(&W[(long)(k0 + 2 * TILE_K + (tid & 31)) * N + colBase], 0, 3);
        }
        __syncthreads();

        // ---- A fragment: two aligned 16B LDS loads per lane ----
        AFrag af;
        {
            int m  = (wave << 4) + (lane & 15);
            int kb = (lane >> 4) << 4;        // 0 or 16 halves
            af.h[0] = *(const v8h*)&As[m][kb];
            af.h[1] = *(const v8h*)&As[m][kb + 8];
        }
        #pragma unroll
        for (int j = 0; j < 4; ++j) {
            AFrag bf;
            bf.h[0] = *(const v8h*)&Bsw[j][lane][0];
            bf.h[1] = *(const v8h*)&Bsw[j][lane][8];
            acc[j] = __builtin_amdgcn_wmma_f32_16x16x32_f16(
                false, af.v, false, bf.v, (short)0, acc[j], false, false);
        }
        __syncthreads();

        if (more) {
            #pragma unroll
            for (int t = 0; t < 4; ++t) { aCur[t] = aNxt[t]; bCur[t] = bNxt[t]; }
        }
    }

    // D layout: VGPR v -> row v (lanes 0-15) or v+8 (lanes 16-31); col = lane%16
    #pragma unroll
    for (int j = 0; j < 4; ++j) {
        int n = colBase + (j << 4) + (lane & 15);
        #pragma unroll
        for (int v = 0; v < 8; ++v) {
            int m = rowBase + (wave << 4) + ((lane >> 4) << 3) + v;
            if (m < M && n < N)
                C[(long)m * N + n] = acc[j][v] + bias[n];
        }
    }
}

// One wave per edge: d2 = ||h[src]-h[dst]||^2, w = exp(-gamma*d2),
// fused scatter: num[dst] += w*h[src], denom[dst] += w.  Requires C % 4 == 0.
__global__ __launch_bounds__(256) void edge_gauss_scatter(
    const float* __restrict__ h, const int* __restrict__ ei,
    const float* __restrict__ gamma_p,
    float* __restrict__ num, float* __restrict__ denom,
    int E, int C)
{
    int e = blockIdx.x * 8 + (threadIdx.x >> 5);
    if (e >= E) return;
    int lane = threadIdx.x & 31;
    int src = ei[e];
    int dst = ei[E + e];
    const float4* hs4 = (const float4*)(h + (long)src * C);
    const float4* hd4 = (const float4*)(h + (long)dst * C);
    const int C4 = C >> 2;
    float d2 = 0.0f;
    for (int c4 = lane; c4 < C4; c4 += 32) {
        float4 a = hs4[c4], b = hd4[c4];
        float dx = a.x - b.x, dy = a.y - b.y, dz = a.z - b.z, dw = a.w - b.w;
        d2 += dx * dx + dy * dy + dz * dz + dw * dw;
    }
    #pragma unroll
    for (int off = 16; off > 0; off >>= 1)
        d2 += __shfl_xor(d2, off, 32);
    float w = expf(-gamma_p[0] * d2);
    float* np = num + (long)dst * C;
    for (int c4 = lane; c4 < C4; c4 += 32) {
        float4 a = hs4[c4];
        atomicAdd(&np[4 * c4 + 0], w * a.x);
        atomicAdd(&np[4 * c4 + 1], w * a.y);
        atomicAdd(&np[4 * c4 + 2], w * a.z);
        atomicAdd(&np[4 * c4 + 3], w * a.w);
    }
    if (lane == 0) atomicAdd(&denom[dst], w);
}

// x = num/(denom+eps) written in place; per-channel partial sum/sumsq,
// one atomic per (block, channel).
__global__ __launch_bounds__(256) void normalize_stats(
    float* __restrict__ num, const float* __restrict__ denom,
    float* __restrict__ sum, float* __restrict__ sumsq,
    int Nn, int C)
{
    const int NPB = 128;
    int c = threadIdx.x;
    if (c >= C) return;
    int n0 = blockIdx.x * NPB;
    int n1 = n0 + NPB; if (n1 > Nn) n1 = Nn;
    float s = 0.0f, ss = 0.0f;
    for (int n = n0; n < n1; ++n) {
        float x = num[(long)n * C + c] / (denom[n] + 1e-16f);
        num[(long)n * C + c] = x;
        s += x; ss += x * x;
    }
    atomicAdd(&sum[c], s);
    atomicAdd(&sumsq[c], ss);
}

__global__ void bn_prep(const float* __restrict__ sum, const float* __restrict__ sumsq,
                        const float* __restrict__ g, const float* __restrict__ b,
                        float* __restrict__ scale, float* __restrict__ shift,
                        int Nn, int C)
{
    int c = threadIdx.x + blockIdx.x * blockDim.x;
    if (c >= C) return;
    float invN = 1.0f / (float)Nn;
    float mean = sum[c] * invN;
    float var  = sumsq[c] * invN - mean * mean;
    float inv  = rsqrtf(var + 1e-5f);
    float sc   = inv * g[c];
    scale[c] = sc;
    shift[c] = b[c] - mean * sc;
}

// y = relu(x*scale[c] + shift[c]), in place safe.
__global__ __launch_bounds__(256) void bn_relu(
    float* __restrict__ x, const float* __restrict__ scale,
    const float* __restrict__ shift, long total, int C)
{
    long i = (long)blockIdx.x * blockDim.x + threadIdx.x;
    if (i >= total) return;
    int c = (int)(i % C);
    float v = x[i] * scale[c] + shift[c];
    x[i] = fmaxf(v, 0.0f);
}

// One wave per batch row: gather + log_softmax over C (<=64).
__global__ __launch_bounds__(256) void gather_logsoftmax(
    const float* __restrict__ x, const int* __restrict__ idx,
    float* __restrict__ out, int B, int C)
{
    int i = blockIdx.x * 8 + (threadIdx.x >> 5);
    if (i >= B) return;
    int lane = threadIdx.x & 31;
    int row  = idx[i];
    const float* xp = x + (long)row * C;
    float ninf = -__builtin_inff();
    float v0 = (lane      < C) ? xp[lane]      : ninf;
    float v1 = (lane + 32 < C) ? xp[lane + 32] : ninf;
    float m = fmaxf(v0, v1);
    #pragma unroll
    for (int off = 16; off > 0; off >>= 1)
        m = fmaxf(m, __shfl_xor(m, off, 32));
    float s = ((lane      < C) ? expf(v0 - m) : 0.0f)
            + ((lane + 32 < C) ? expf(v1 - m) : 0.0f);
    #pragma unroll
    for (int off = 16; off > 0; off >>= 1)
        s += __shfl_xor(s, off, 32);
    float lse = m + logf(s);
    if (lane      < C) out[(long)i * C + lane]      = v0 - lse;
    if (lane + 32 < C) out[(long)i * C + lane + 32] = v1 - lse;
}

extern "C" void kernel_launch(void* const* d_in, const int* in_sizes, int n_in,
                              void* d_out, int out_size, void* d_ws, size_t ws_size,
                              hipStream_t stream) {
    const float* features   = (const float*)d_in[0];
    const int*   edge_index = (const int*)d_in[1];
    const int*   batch_nodes= (const int*)d_in[2];
    const float* W1     = (const float*)d_in[3];
    const float* b1     = (const float*)d_in[4];
    const float* gamma1 = (const float*)d_in[5];
    const float* W2     = (const float*)d_in[6];
    const float* b2     = (const float*)d_in[7];
    const float* gamma2 = (const float*)d_in[8];
    const float* bn1_w  = (const float*)d_in[9];
    const float* bn1_b  = (const float*)d_in[10];
    const float* bn2_w  = (const float*)d_in[11];
    const float* bn2_b  = (const float*)d_in[12];

    const int H  = in_sizes[4];              // 256
    const int F  = in_sizes[3] / H;          // 256
    const int Nn = in_sizes[0] / F;          // 50000
    const int E  = in_sizes[1] / 2;          // 800000
    const int O  = in_sizes[7];              // 40
    const int B  = in_sizes[2];              // 10000

    // Workspace layout (floats)
    float* ws     = (float*)d_ws;
    float* h1     = ws;                         // Nn*H
    float* num1   = h1    + (size_t)Nn * H;     // Nn*H (becomes x1 in place)
    float* denom1 = num1  + (size_t)Nn * H;     // Nn
    float* sum1   = denom1 + Nn;                // H
    float* sumsq1 = sum1   + H;                 // H
    float* scale1 = sumsq1 + H;                 // H
    float* shift1 = scale1 + H;                 // H
    float* h2     = shift1 + H;                 // Nn*O
    float* num2   = h2    + (size_t)Nn * O;     // Nn*O (becomes x2 in place)
    float* denom2 = num2  + (size_t)Nn * O;     // Nn
    float* sum2   = denom2 + Nn;                // O
    float* sumsq2 = sum2   + O;                 // O
    float* scale2 = sumsq2 + O;                 // O
    float* shift2 = scale2 + O;                 // O

    // Zero accumulators (graph-capturable memsets)
    hipMemsetAsync(num1,   0, (size_t)Nn * H * sizeof(float), stream);
    hipMemsetAsync(denom1, 0, (size_t)Nn * sizeof(float), stream);
    hipMemsetAsync(sum1,   0, 2 * H * sizeof(float), stream);   // sum1+sumsq1
    hipMemsetAsync(num2,   0, (size_t)Nn * O * sizeof(float), stream);
    hipMemsetAsync(denom2, 0, (size_t)Nn * sizeof(float), stream);
    hipMemsetAsync(sum2,   0, 2 * O * sizeof(float), stream);   // sum2+sumsq2

    dim3 gblk(128);

    // ---- Layer 1 ----
    {
        dim3 grid((Nn + TILE_M - 1) / TILE_M, (H + TILE_N - 1) / TILE_N);
        gemm_bias_wmma<<<grid, gblk, 0, stream>>>(features, W1, b1, h1, Nn, F, H);
    }
    edge_gauss_scatter<<<(E + 7) / 8, 256, 0, stream>>>(
        h1, edge_index, gamma1, num1, denom1, E, H);
    normalize_stats<<<(Nn + 127) / 128, 256, 0, stream>>>(
        num1, denom1, sum1, sumsq1, Nn, H);
    bn_prep<<<(H + 255) / 256, 256, 0, stream>>>(
        sum1, sumsq1, bn1_w, bn1_b, scale1, shift1, Nn, H);
    {
        long total = (long)Nn * H;
        bn_relu<<<(int)((total + 255) / 256), 256, 0, stream>>>(
            num1, scale1, shift1, total, H);
    }

    // ---- Layer 2 ----
    {
        dim3 grid((Nn + TILE_M - 1) / TILE_M, (O + TILE_N - 1) / TILE_N);
        gemm_bias_wmma<<<grid, gblk, 0, stream>>>(num1, W2, b2, h2, Nn, H, O);
    }
    edge_gauss_scatter<<<(E + 7) / 8, 256, 0, stream>>>(
        h2, edge_index, gamma2, num2, denom2, E, O);
    normalize_stats<<<(Nn + 127) / 128, 256, 0, stream>>>(
        num2, denom2, sum2, sumsq2, Nn, O);
    bn_prep<<<(O + 255) / 256, 256, 0, stream>>>(
        sum2, sumsq2, bn2_w, bn2_b, scale2, shift2, Nn, O);
    {
        long total = (long)Nn * O;
        bn_relu<<<(int)((total + 255) / 256), 256, 0, stream>>>(
            num2, scale2, shift2, total, O);
    }

    // ---- Gather + log_softmax ----
    gather_logsoftmax<<<(B + 7) / 8, 256, 0, stream>>>(
        num2, batch_nodes, (float*)d_out, B, O);
}
